// GCNNet_9208409883138
// MI455X (gfx1250) — matrix operand order repaired and checked by reading, not compiled, not run on hardware
//
#include <hip/hip_runtime.h>
#include <hip/hip_bf16.h>

typedef __attribute__((ext_vector_type(16))) _Float16 v16h;
typedef __attribute__((ext_vector_type(8)))  float    v8f;

#define NF   128
#define NCLS 10
#define NG   64

// ---------------------------------------------------------------------------
// Degree: deg[dst] += 1 per edge (native f32 atomics; deg buffer pre-zeroed)
// ---------------------------------------------------------------------------
__global__ void degree_kernel(const int* __restrict__ dst, float* __restrict__ deg, int E) {
    int e = blockIdx.x * blockDim.x + threadIdx.x;
    if (e < E) unsafeAtomicAdd(&deg[dst[e]], 1.0f);
}

// dinv[n] = rsqrt(deg[n] + 1)  (in place)
__global__ void dinv_kernel(float* __restrict__ deg, int N) {
    int n = blockIdx.x * blockDim.x + threadIdx.x;
    if (n < N) deg[n] = rsqrtf(deg[n] + 1.0f);
}

// ---------------------------------------------------------------------------
// Weight transpose + f32->f16 convert: Wt[n][k] = (f16) W[k][n].
// Makes each lane's WMMA B-fragment (16 consecutive K for one column) a single
// contiguous 32-byte load. 128x128, launched once per layer (negligible).
// ---------------------------------------------------------------------------
__global__ void wcvt_kernel(const float* __restrict__ W, _Float16* __restrict__ Wt) {
    int i = blockIdx.x * blockDim.x + threadIdx.x;   // 16384 threads
    int k = i >> 7, n = i & (NF - 1);
    Wt[(size_t)n * NF + k] = (_Float16)W[(size_t)k * NF + n];
}

// ---------------------------------------------------------------------------
// WMMA GEMM: H[N,128] = X[N,128] @ W[128,128]   (f32 X, f16 WMMA, f32 acc)
// block = 256 threads = 8 waves; wave w owns the 16x16 tile at cols w*16,
// block b owns rows b*16..b*16+15. K loop: 4 x WMMA_F32_16X16X32_F16.
// ---------------------------------------------------------------------------
__global__ void __launch_bounds__(256)
gemm_wmma(const float* __restrict__ X, const _Float16* __restrict__ Wt,
          float* __restrict__ H) {
    const int wave = threadIdx.x >> 5;
    const int lane = threadIdx.x & 31;
    const int row  = lane & 15;      // M index within tile / N index for B,C
    const int grp  = lane >> 4;      // half-wave group
    const int m0   = blockIdx.x * 16;
    const int n0   = wave * 16;

    const float* __restrict__ xrow = X + (size_t)(m0 + row) * NF;
    const int    col               = n0 + row;
    const _Float16* __restrict__ wcol = Wt + (size_t)col * NF + grp * 16;

    v8f acc = {};
#pragma unroll
    for (int k0 = 0; k0 < NF; k0 += 32) {
        v16h a, b;
        // A fragment (16x32 f16): lanes 0-15: halves 0-15 -> K 0-15
        //                         lanes 16-31: halves 0-15 -> K 16-31
#pragma unroll
        for (int j = 0; j < 8; ++j) {
            a[j]     = (_Float16)xrow[k0 + grp * 16 + j];
            a[j + 8] = (_Float16)xrow[k0 + grp * 16 + 8 + j];
        }
        // B fragment (32x16 f16): lane holds column (lane&15), 16 consecutive K
        // starting at k0 + (lane>>4)*16 -> one contiguous 32B vector load
        b = *(const v16h*)(wcol + k0);
        acc = __builtin_amdgcn_wmma_f32_16x16x32_f16(
            /*neg_a=*/false, a, /*neg_b=*/false, b,
            /*c_mod=*/(short)0, acc, /*reuse_a=*/false, /*reuse_b=*/false);
    }
    // C/D layout: element v of lane -> M = v + 8*grp, N = lane&15
    const int cm = m0 + grp * 8;
#pragma unroll
    for (int v = 0; v < 8; ++v)
        H[(size_t)(cm + v) * NF + col] = acc[v];
}

// ---------------------------------------------------------------------------
// Edge scatter: one wave per edge; lane handles 4 features (32*4 = 128).
// AGG[dst] += H[src] * dinv[src]*dinv[dst]  (native global_atomic_add_f32;
// entire node buffer is L2-resident so these are L2 atomics)
// ---------------------------------------------------------------------------
__global__ void __launch_bounds__(256)
scatter_kernel(const float* __restrict__ H, const int* __restrict__ src,
               const int* __restrict__ dst, const float* __restrict__ dinv,
               float* __restrict__ AGG, int E) {
    int tid  = blockIdx.x * blockDim.x + threadIdx.x;
    int wid  = tid >> 5;
    int lane = threadIdx.x & 31;
    if (wid >= E) return;
    int s = src[wid], d = dst[wid];
    float norm = dinv[s] * dinv[d];
    const float4* hs = (const float4*)(H + (size_t)s * NF);
    float4 hv = hs[lane];
    float* out = AGG + (size_t)d * NF + lane * 4;
    unsafeAtomicAdd(out + 0, hv.x * norm);
    unsafeAtomicAdd(out + 1, hv.y * norm);
    unsafeAtomicAdd(out + 2, hv.z * norm);
    unsafeAtomicAdd(out + 3, hv.w * norm);
}

// ---------------------------------------------------------------------------
// Finalize: out = agg + h*dinv^2 + bias (+ReLU); written in place into AGG.
// ---------------------------------------------------------------------------
__global__ void finalize_kernel(const float* __restrict__ H, float* __restrict__ AGG,
                                const float* __restrict__ dinv,
                                const float* __restrict__ bias, int relu, int total) {
    int i = blockIdx.x * blockDim.x + threadIdx.x;
    if (i >= total) return;
    int n = i >> 7, f = i & (NF - 1);
    float di = dinv[n];
    float v = AGG[i] + H[i] * di * di + bias[f];
    if (relu) v = fmaxf(v, 0.0f);
    AGG[i] = v;
}

// ---------------------------------------------------------------------------
// Global max pool via order-preserving uint encoding + atomicMax.
// enc of any real float > 0, so memset(0) init is the identity element.
// ---------------------------------------------------------------------------
__device__ __forceinline__ unsigned enc_f(float x) {
    unsigned b = __float_as_uint(x);
    return (b & 0x80000000u) ? ~b : (b | 0x80000000u);
}
__device__ __forceinline__ float dec_f(unsigned e) {
    unsigned b = (e & 0x80000000u) ? (e & 0x7FFFFFFFu) : ~e;
    return __uint_as_float(b);
}

__global__ void pool_kernel(const float* __restrict__ H, const int* __restrict__ batch,
                            unsigned* __restrict__ genc, int total) {
    int i = blockIdx.x * blockDim.x + threadIdx.x;
    if (i >= total) return;
    int n = i >> 7, f = i & (NF - 1);
    atomicMax(&genc[(size_t)batch[n] * NF + f], enc_f(H[i]));
}

// ---------------------------------------------------------------------------
// Head: g(64x128) -> fc1(128x64)+ReLU -> fc2(64x10) -> log_softmax. One block
// per graph, 64 threads. Tiny; scalar is fine.
// ---------------------------------------------------------------------------
__global__ void __launch_bounds__(64)
head_kernel(const unsigned* __restrict__ genc,
            const float* __restrict__ fc1_w, const float* __restrict__ fc1_b,
            const float* __restrict__ fc2_w, const float* __restrict__ fc2_b,
            float* __restrict__ out) {
    __shared__ float gs[NF];
    __shared__ float t1[64];
    __shared__ float t2[NCLS];
    int g = blockIdx.x, t = threadIdx.x;
    gs[t]      = dec_f(genc[(size_t)g * NF + t]);
    gs[t + 64] = dec_f(genc[(size_t)g * NF + t + 64]);
    __syncthreads();
    float s = fc1_b[t];
#pragma unroll 8
    for (int k = 0; k < NF; ++k) s += gs[k] * fc1_w[(size_t)k * 64 + t];
    t1[t] = fmaxf(s, 0.0f);
    __syncthreads();
    if (t < NCLS) {
        float s2 = fc2_b[t];
#pragma unroll 8
        for (int k = 0; k < 64; ++k) s2 += t1[k] * fc2_w[(size_t)k * NCLS + t];
        t2[t] = s2;
    }
    __syncthreads();
    if (t < NCLS) {
        float m = t2[0];
#pragma unroll
        for (int k = 1; k < NCLS; ++k) m = fmaxf(m, t2[k]);
        float sum = 0.0f;
#pragma unroll
        for (int k = 0; k < NCLS; ++k) sum += __expf(t2[k] - m);
        out[(size_t)g * NCLS + t] = t2[t] - m - __logf(sum);
    }
}

// ---------------------------------------------------------------------------
extern "C" void kernel_launch(void* const* d_in, const int* in_sizes, int n_in,
                              void* d_out, int out_size, void* d_ws, size_t ws_size,
                              hipStream_t stream) {
    const float* x     = (const float*)d_in[0];
    const int*   ei    = (const int*)d_in[1];
    const int*   batch = (const int*)d_in[2];
    const float* W1 = (const float*)d_in[3];
    const float* b1 = (const float*)d_in[4];
    const float* W2 = (const float*)d_in[5];
    const float* b2 = (const float*)d_in[6];
    const float* W3 = (const float*)d_in[7];
    const float* b3 = (const float*)d_in[8];
    const float* fc1_w = (const float*)d_in[9];
    const float* fc1_b = (const float*)d_in[10];
    const float* fc2_w = (const float*)d_in[11];
    const float* fc2_b = (const float*)d_in[12];
    float* out = (float*)d_out;

    const int N = in_sizes[0] / NF;     // 50000
    const int E = in_sizes[1] / 2;      // 1600000
    const int* src = ei;
    const int* dst = ei + E;

    const size_t nodeElems = (size_t)N * NF;
    char* ws = (char*)d_ws;
    float* B0   = (float*)ws;                            ws += nodeElems * sizeof(float);
    float* B1   = (float*)ws;                            ws += nodeElems * sizeof(float);
    float* B2   = (float*)ws;                            ws += nodeElems * sizeof(float);
    float* dinv = (float*)ws;                            ws += (size_t)N * sizeof(float);
    unsigned* genc = (unsigned*)ws;                      ws += (size_t)NG * NF * sizeof(unsigned);
    _Float16* Wh = (_Float16*)ws;                        ws += (size_t)NF * NF * sizeof(_Float16);

    const int total = N * NF;
    const int gBlocks = N / 16;                // N divisible by 16
    const int eThreadBlocks = (E + 255) / 256;
    const int eWaveBlocks   = (int)(((size_t)E * 32 + 255) / 256);
    const int nfBlocks = (total + 255) / 256;
    const int wBlocks  = (NF * NF) / 256;      // 64

    // degree -> dinv
    hipMemsetAsync(dinv, 0, (size_t)N * sizeof(float), stream);
    degree_kernel<<<eThreadBlocks, 256, 0, stream>>>(dst, dinv, E);
    dinv_kernel<<<(N + 255) / 256, 256, 0, stream>>>(dinv, N);

    // Layer 1: in=x -> H=B0 -> AGG=B1 (out1 in B1, ReLU)
    wcvt_kernel<<<wBlocks, 256, 0, stream>>>(W1, Wh);
    gemm_wmma<<<gBlocks, 256, 0, stream>>>(x, Wh, B0);
    hipMemsetAsync(B1, 0, nodeElems * sizeof(float), stream);
    scatter_kernel<<<eWaveBlocks, 256, 0, stream>>>(B0, src, dst, dinv, B1, E);
    finalize_kernel<<<nfBlocks, 256, 0, stream>>>(B0, B1, dinv, b1, 1, total);

    // Layer 2: in=B1 -> H=B0 -> AGG=B2 (out2 in B2, ReLU)
    wcvt_kernel<<<wBlocks, 256, 0, stream>>>(W2, Wh);
    gemm_wmma<<<gBlocks, 256, 0, stream>>>(B1, Wh, B0);
    hipMemsetAsync(B2, 0, nodeElems * sizeof(float), stream);
    scatter_kernel<<<eWaveBlocks, 256, 0, stream>>>(B0, src, dst, dinv, B2, E);
    finalize_kernel<<<nfBlocks, 256, 0, stream>>>(B0, B2, dinv, b2, 1, total);

    // Layer 3: in=B2 -> H=B0 -> AGG=B1 (out3 in B1, no ReLU)
    wcvt_kernel<<<wBlocks, 256, 0, stream>>>(W3, Wh);
    gemm_wmma<<<gBlocks, 256, 0, stream>>>(B2, Wh, B0);
    hipMemsetAsync(B1, 0, nodeElems * sizeof(float), stream);
    scatter_kernel<<<eWaveBlocks, 256, 0, stream>>>(B0, src, dst, dinv, B1, E);
    finalize_kernel<<<nfBlocks, 256, 0, stream>>>(B0, B1, dinv, b3, 0, total);

    // Global max pool + head
    hipMemsetAsync(genc, 0, (size_t)NG * NF * sizeof(unsigned), stream);
    pool_kernel<<<nfBlocks, 256, 0, stream>>>(B1, batch, genc, total);
    head_kernel<<<NG, 64, 0, stream>>>(genc, fc1_w, fc1_b, fc2_w, fc2_b, out);
}
